// GRNNEncoder_20701742367276
// MI455X (gfx1250) — compile-verified, matrix operand order checked by reading, not run on hardware
//
// NRI graph-recurrent encoder for MI455X (gfx1250), bf16 WMMA everywhere.
// GEMM: 64x64 block tile, 8 waves (4Mx2N), each wave a 16x32 strip (2 WMMA acc,
// A-fragment reused across both) -> 3 ds_load_b128 per v_wmma_f32_16x16x32_bf16.
#include <hip/hip_runtime.h>
#include <math.h>

typedef __bf16 bf16;
typedef bf16  v16bf __attribute__((ext_vector_type(16)));
typedef bf16  v8bf  __attribute__((ext_vector_type(8)));
typedef float v8f   __attribute__((ext_vector_type(8)));

#define B_    8
#define N_    64
#define T_    50
#define D_    4
#define R_    4032            // N*(N-1), edges enumerated recv-major
#define MROWS (B_ * R_)       // 32256
#define MAXK  192
#define SPAD  8               // +16B LDS row padding (keeps 16B alignment)

__device__ __forceinline__ float sigmoidf_(float x) { return 1.f / (1.f + __expf(-x)); }
__device__ __forceinline__ float eluf_(float x)     { return x > 0.f ? x : __expf(x) - 1.f; }

// A/B fragment load matching CDNA5 16-bit WMMA VGPR layout:
// row = lane%16; vector elems 0..7 -> K = k0 + 8*(lane/16) + [0..7],
// elems 8..15 -> same + 16. Two 16-byte LDS loads per fragment.
__device__ __forceinline__ v16bf load_frag(const bf16* base, int stride, int k0, int lane) {
  const bf16* p = base + (lane & 15) * stride + k0 + ((lane >> 4) << 3);
  v8bf lo = *(const v8bf*)p;
  v8bf hi = *(const v8bf*)(p + 16);
  return __builtin_shufflevector(lo, hi, 0,1,2,3,4,5,6,7,8,9,10,11,12,13,14,15);
}

enum { EPI_F32 = 0, EPI_ELU_BF16 = 1, EPI_ELU_F32_STATS = 2, EPI_ELU_F32 = 3 };

// Y[M,Ntot] = epi(A[M,K](bf16) @ W[Ntot,K]^T(bf16) + bias).
template<int EPI>
__global__ void __launch_bounds__(256) gemm_bf16_kernel(
    const bf16* __restrict__ A, int lda,
    const bf16* __restrict__ W, int ldw,
    const float* __restrict__ bias,
    int K, int ldy,
    float* __restrict__ Yf, bf16* __restrict__ Yb,
    float* __restrict__ stats /* [2][64] running sums for BN */) {
  __shared__ alignas(16) bf16 As[64 * (MAXK + SPAD)];
  __shared__ alignas(16) bf16 Bs[64 * (MAXK + SPAD)];
  const int SP  = K + SPAD;
  const int m0  = blockIdx.x * 64;
  const int n0  = blockIdx.y * 64;
  const int tid = threadIdx.x;
  const int kc8 = K >> 3;
  for (int idx = tid; idx < 64 * kc8; idx += 256) {
    int row = idx / kc8, kc = idx - row * kc8;
    float4 v = *(const float4*)(A + (size_t)(m0 + row) * lda + (kc << 3));
    *(float4*)(&As[row * SP + (kc << 3)]) = v;
  }
  for (int idx = tid; idx < 64 * kc8; idx += 256) {
    int row = idx / kc8, kc = idx - row * kc8;
    float4 v = *(const float4*)(W + (size_t)(n0 + row) * ldw + (kc << 3));
    *(float4*)(&Bs[row * SP + (kc << 3)]) = v;
  }
  __syncthreads();

  const int wave = tid >> 5, lane = tid & 31;
  const int wm = wave & 3, wn = wave >> 2;       // 4x2 wave grid -> 64x64 tile
  v8f acc0 = {0.f, 0.f, 0.f, 0.f, 0.f, 0.f, 0.f, 0.f};
  v8f acc1 = {0.f, 0.f, 0.f, 0.f, 0.f, 0.f, 0.f, 0.f};
  for (int k0 = 0; k0 < K; k0 += 32) {
    v16bf af = load_frag(&As[(wm * 16) * SP], SP, k0, lane);
    v16bf b0 = load_frag(&Bs[(wn * 32) * SP], SP, k0, lane);
    v16bf b1 = load_frag(&Bs[(wn * 32 + 16) * SP], SP, k0, lane);
    acc0 = __builtin_amdgcn_wmma_f32_16x16x32_bf16(false, af, false, b0,
                                                   (short)0, acc0, false, false);
    acc1 = __builtin_amdgcn_wmma_f32_16x16x32_bf16(false, af, false, b1,
                                                   (short)0, acc1, false, false);
  }

  // C/D layout: lane holds column n = lane%16; elem v -> row wm*16 + v + 8*(lane/16)
  const int n_lo  = n0 + wn * 32 + (lane & 15);
  const int n_hi  = n_lo + 16;
  const int mbase = m0 + wm * 16 + ((lane >> 4) << 3);
  const float bv0 = bias ? bias[n_lo] : 0.f;
  const float bv1 = bias ? bias[n_hi] : 0.f;
  float lsum0 = 0.f, lsq0 = 0.f, lsum1 = 0.f, lsq1 = 0.f;
#pragma unroll
  for (int v = 0; v < 8; ++v) {
    float y0 = acc0[v] + bv0;
    float y1 = acc1[v] + bv1;
    if (EPI != EPI_F32) { y0 = eluf_(y0); y1 = eluf_(y1); }
    const size_t o = (size_t)(mbase + v) * ldy;
    if (EPI == EPI_ELU_BF16) { Yb[o + n_lo] = (bf16)y0; Yb[o + n_hi] = (bf16)y1; }
    else                     { Yf[o + n_lo] = y0;       Yf[o + n_hi] = y1; }
    lsum0 += y0; lsq0 += y0 * y0;
    lsum1 += y1; lsq1 += y1 * y1;
  }
  if (EPI == EPI_ELU_F32_STATS) {
    atomicAdd(&stats[n_lo & 63], lsum0);
    atomicAdd(&stats[64 + (n_lo & 63)], lsq0);
    atomicAdd(&stats[n_hi & 63], lsum1);
    atomicAdd(&stats[64 + (n_hi & 63)], lsq1);
  }
}

__global__ void zero_f32_kernel(float* p, size_t n) {
  size_t i = (size_t)blockIdx.x * 256 + threadIdx.x;
  size_t st = (size_t)gridDim.x * 256;
  for (; i < n; i += st) p[i] = 0.f;
}

__global__ void cvt_bf16_kernel(bf16* dst, const float* src, int n) {
  int i = blockIdx.x * 256 + threadIdx.x;
  if (i < n) dst[i] = (bf16)src[i];
}

// Wg[g][k] = k<kx ? wih[g,k] : whh[g,k-kx]  (bf16), g in [0,256)
__global__ void prep_gatew_kernel(bf16* dst, const float* wih, const float* whh, int kx) {
  int K = kx + 64;
  int idx = blockIdx.x * 256 + threadIdx.x;
  if (idx >= 256 * K) return;
  int g = idx / K, k = idx - g * K;
  dst[idx] = (bf16)((k < kx) ? wih[g * kx + k] : whh[g * 64 + (k - kx)]);
}

__global__ void add_bias_kernel(float* dst, const float* a, const float* b, int n) {
  int i = blockIdx.x * 256 + threadIdx.x;
  if (i < n) dst[i] = a[i] + b[i];
}

__global__ void finalize_stats_kernel(float* stats, float* mu, float* rsig, float invcnt) {
  int c = threadIdx.x;  // 64 threads
  float s = stats[c], sq = stats[64 + c];
  float m = s * invcnt;
  float var = sq * invcnt - m * m;
  mu[c] = m;
  rsig[c] = rsqrtf(var + 1e-5f);
  stats[c] = 0.f; stats[64 + c] = 0.f;
}

// mlp1 fc1 (K=4, trivially small): row = (b*T+t)*N + n of x=[B*T,N,D]
__global__ void mlp1_fc1_kernel(const float* __restrict__ in, const float* __restrict__ w1,
                                const float* __restrict__ b1, bf16* __restrict__ t1) {
  int idx = blockIdx.x * 256 + threadIdx.x;          // 25600*64
  if (idx >= B_ * T_ * N_ * 64) return;
  int oc = idx & 63; int row = idx >> 6;
  int n = row % N_; int bt = row / N_; int t = bt % T_; int b = bt / T_;
  const float* xp = in + (((size_t)(b * N_ + n)) * T_ + t) * D_;
  float a = w1[oc*4+0]*xp[0] + w1[oc*4+1]*xp[1] + w1[oc*4+2]*xp[2] + w1[oc*4+3]*xp[3] + b1[oc];
  t1[idx] = (bf16)eluf_(a);
}

// node2edge from mlp1 output (already normalized, bf16): E[b*R+e][0:64]=sender, [64:128]=receiver
__global__ void gather_x_kernel(const bf16* __restrict__ xn, int t, bf16* __restrict__ E) {
  int idx = blockIdx.x * 256 + threadIdx.x;          // MROWS*128
  if (idx >= MROWS * 128) return;
  int c = idx & 127; int be = idx >> 7;
  int b = be / R_; int e = be - b * R_;
  int n = e / 63, j = e - n * 63;
  int node = (c < 64) ? (j + (j >= n ? 1 : 0)) : n;  // send : recv
  E[idx] = xn[(((size_t)b * T_ + t) * N_ + node) * 64 + (c & 63)];
}

// node2edge from raw node features with BN affine applied on the fly
__global__ void gather_bn_kernel(const float* __restrict__ nodes,
                                 const float* __restrict__ mu, const float* __restrict__ rsig,
                                 const float* __restrict__ gamma, const float* __restrict__ beta,
                                 bf16* __restrict__ E) {
  int idx = blockIdx.x * 256 + threadIdx.x;          // MROWS*128
  if (idx >= MROWS * 128) return;
  int c = idx & 127; int be = idx >> 7;
  int b = be / R_; int e = be - b * R_;
  int n = e / 63, j = e - n * 63;
  int node = (c < 64) ? (j + (j >= n ? 1 : 0)) : n;
  int ch = c & 63;
  float v = nodes[((size_t)b * N_ + node) * 64 + ch];
  v = (v - mu[ch]) * rsig[ch] * gamma[ch] + beta[ch];
  E[idx] = (bf16)v;
}

// gates[r,0:256]=[i|f|g|o] -> c,h update
__global__ void lstm_cell_kernel(const float* __restrict__ gates,
                                 float* __restrict__ c, float* __restrict__ hl) {
  int idx = blockIdx.x * 256 + threadIdx.x;          // MROWS*64
  if (idx >= MROWS * 64) return;
  int r = idx >> 6, h = idx & 63;
  const float* g = gates + (size_t)r * 256;
  float ig = sigmoidf_(g[h]);
  float fg = sigmoidf_(g[64 + h]);
  float gg = tanhf(g[128 + h]);
  float og = sigmoidf_(g[192 + h]);
  float cn = fg * c[idx] + ig * gg;
  c[idx] = cn;
  hl[idx] = og * tanhf(cn);
}

// recv-major edges: node n receives edges [n*63, n*63+63); mean divides by N=64
__global__ void edge2node_kernel(const float* __restrict__ hl, bf16* __restrict__ node_in) {
  int idx = blockIdx.x * 256 + threadIdx.x;          // B*N*64 = 32768
  if (idx >= B_ * N_ * 64) return;
  int ch = idx & 63; int bn = idx >> 6; int b = bn >> 6; int n = bn & 63;
  const float* base = hl + ((size_t)b * R_ + n * 63) * 64 + ch;
  float s = 0.f;
  for (int j = 0; j < 63; ++j) s += base[(size_t)j * 64];
  node_in[idx] = (bf16)(s * (1.f / 64.f));
}

// BN affine, fan-out bf16 writes into up to three packed GEMM-input slots (+opt f32)
__global__ void bn_write_kernel(const float* __restrict__ raw,
                                const float* __restrict__ mu, const float* __restrict__ rsig,
                                const float* __restrict__ gamma, const float* __restrict__ beta,
                                bf16* d0, int ld0, bf16* d1, int ld1, bf16* d2, int ld2,
                                float* df, int rows) {
  int idx = blockIdx.x * 256 + threadIdx.x;
  if (idx >= rows * 64) return;
  int ch = idx & 63; int r = idx >> 6;
  float v = (raw[idx] - mu[ch]) * rsig[ch] * gamma[ch] + beta[ch];
  bf16 bv = (bf16)v;
  if (d0) d0[(size_t)r * ld0 + ch] = bv;
  if (d1) d1[(size_t)r * ld1 + ch] = bv;
  if (d2) d2[(size_t)r * ld2 + ch] = bv;
  if (df) df[idx] = v;
}

__global__ void out_kernel(const float* __restrict__ h2, const float* __restrict__ ow,
                           const float* __restrict__ ob, float* __restrict__ out) {
  int idx = blockIdx.x * 256 + threadIdx.x;          // MROWS*2
  if (idx >= MROWS * 2) return;
  int e = idx & 1; int r = idx >> 1;
  const float* h = h2 + (size_t)r * 64;
  const float* w = ow + e * 64;
  float s = ob[e];
  for (int k = 0; k < 64; ++k) s += h[k] * w[k];
  out[idx] = s;
}

// ---- param indices (setup_inputs dict order, nested dicts flattened) ----
enum {
  I_INPUTS = 0, I_RELREC, I_RELSEND,
  I_M1W1 = 3, I_M1B1, I_M1W2, I_M1B2, I_M1G, I_M1B,
  I_M2W1 = 9, I_M2B1, I_M2W2, I_M2B2, I_M2G, I_M2B,
  I_M3W1 = 15, I_M3B1, I_M3W2, I_M3B2, I_M3G, I_M3B,
  I_M4W1 = 21, I_M4B1, I_M4W2, I_M4B2, I_M4G, I_M4B,
  I_L0WIH = 27, I_L0WHH, I_L0BIH, I_L0BHH,
  I_L1WIH = 31, I_L1WHH, I_L1BIH, I_L1BHH,
  I_L2WIH = 35, I_L2WHH, I_L2BIH, I_L2BHH,
  I_OUTW = 39, I_OUTB = 40
};

#define GEMM(EPI, Ap, lda, Wp, ldw, biasp, M, Ntot, K, ldy, Yfp, Ybp, statsp)            \
  gemm_bf16_kernel<EPI><<<dim3((M) / 64, (Ntot) / 64), 256, 0, stream>>>(                \
      (Ap), (lda), (Wp), (ldw), (biasp), (K), (ldy), (Yfp), (Ybp), (statsp))

#define EW(kern, n, ...) kern<<<((n) + 255) / 256, 256, 0, stream>>>(__VA_ARGS__)

extern "C" void kernel_launch(void* const* d_in, const int* in_sizes, int n_in,
                              void* d_out, int out_size, void* d_ws, size_t ws_size,
                              hipStream_t stream) {
  (void)in_sizes; (void)n_in; (void)out_size; (void)ws_size;
  #define F32IN(i) ((const float*)d_in[(i)])

  char* p = (char*)d_ws;
  auto carve = [&](size_t bytes) -> void* {
    void* r = (void*)p; p += (bytes + 255) & ~(size_t)255; return r;
  };
  // weights (bf16) + biases
  bf16* Wm1b = (bf16*)carve(64 * 64 * 2);
  bf16* Wm2a = (bf16*)carve(64 * 128 * 2);
  bf16* Wm2b = (bf16*)carve(64 * 64 * 2);
  bf16* Wm3a = (bf16*)carve(64 * 64 * 2);
  bf16* Wm3b = (bf16*)carve(64 * 64 * 2);
  bf16* Wm4a = (bf16*)carve(64 * 128 * 2);
  bf16* Wm4b = (bf16*)carve(64 * 64 * 2);
  bf16* Wg0  = (bf16*)carve(256 * 128 * 2);
  bf16* Wg1  = (bf16*)carve(256 * 192 * 2);
  bf16* Wg2  = (bf16*)carve(256 * 192 * 2);
  float* bg0 = (float*)carve(256 * 4);
  float* bg1 = (float*)carve(256 * 4);
  float* bg2 = (float*)carve(256 * 4);
  float* stats  = (float*)carve(128 * 4);   // shared (mlp1 / mlp3 / mlp4)
  float* stats2 = (float*)carve(128 * 4);   // mlp2 (accumulates over all T)
  float* mu2 = (float*)carve(64 * 4);
  float* rs2 = (float*)carve(64 * 4);
  float* mus = (float*)carve(64 * 4);
  float* rss = (float*)carve(64 * 4);
  // activations
  const size_t NR1 = (size_t)B_ * T_ * N_;            // 25600
  bf16*  t1    = (bf16*)carve(NR1 * 64 * 2);
  float* x1raw = (float*)carve(NR1 * 64 * 4);
  bf16*  xn    = (bf16*)carve(NR1 * 64 * 2);
  bf16*  E     = (bf16*)carve((size_t)MROWS * 128 * 2);
  bf16*  tmid  = (bf16*)carve((size_t)MROWS * 64 * 2);
  float* raw   = (float*)carve((size_t)MROWS * 64 * 4);
  bf16*  node_in = (bf16*)carve((size_t)B_ * N_ * 64 * 2);
  float* n3raw   = (float*)carve((size_t)B_ * N_ * 64 * 4);
  float* gates = (float*)carve((size_t)MROWS * 256 * 4);
  bf16*  G0 = (bf16*)carve((size_t)MROWS * 128 * 2);  // [xt | h0_state]
  bf16*  G1 = (bf16*)carve((size_t)MROWS * 192 * 2);  // [xt | h0_new | h1_state]
  bf16*  G2 = (bf16*)carve((size_t)MROWS * 192 * 2);  // [h0_new | h1_new | h2_state]
  float* c0 = (float*)carve((size_t)MROWS * 64 * 4);
  float* c1 = (float*)carve((size_t)MROWS * 64 * 4);
  float* c2 = (float*)carve((size_t)MROWS * 64 * 4);
  float* hl = (float*)carve((size_t)MROWS * 64 * 4);
  float* h2 = (float*)carve((size_t)MROWS * 64 * 4);

  // ---- zero state (every call: deterministic) ----
  zero_f32_kernel<<<64, 256, 0, stream>>>(stats, 128);
  zero_f32_kernel<<<64, 256, 0, stream>>>(stats2, 128);
  zero_f32_kernel<<<4096, 256, 0, stream>>>(c0, (size_t)MROWS * 64 * 3);      // c0,c1,c2 contiguous
  zero_f32_kernel<<<4096, 256, 0, stream>>>((float*)G0, (size_t)MROWS * 128 / 2);
  zero_f32_kernel<<<4096, 256, 0, stream>>>((float*)G1, (size_t)MROWS * 192 / 2);
  zero_f32_kernel<<<4096, 256, 0, stream>>>((float*)G2, (size_t)MROWS * 192 / 2);

  // ---- weight prep ----
  EW(cvt_bf16_kernel, 64 * 64,  Wm1b, F32IN(I_M1W2), 64 * 64);
  EW(cvt_bf16_kernel, 64 * 128, Wm2a, F32IN(I_M2W1), 64 * 128);
  EW(cvt_bf16_kernel, 64 * 64,  Wm2b, F32IN(I_M2W2), 64 * 64);
  EW(cvt_bf16_kernel, 64 * 64,  Wm3a, F32IN(I_M3W1), 64 * 64);
  EW(cvt_bf16_kernel, 64 * 64,  Wm3b, F32IN(I_M3W2), 64 * 64);
  EW(cvt_bf16_kernel, 64 * 128, Wm4a, F32IN(I_M4W1), 64 * 128);
  EW(cvt_bf16_kernel, 64 * 64,  Wm4b, F32IN(I_M4W2), 64 * 64);
  EW(prep_gatew_kernel, 256 * 128, Wg0, F32IN(I_L0WIH), F32IN(I_L0WHH), 64);
  EW(prep_gatew_kernel, 256 * 192, Wg1, F32IN(I_L1WIH), F32IN(I_L1WHH), 128);
  EW(prep_gatew_kernel, 256 * 192, Wg2, F32IN(I_L2WIH), F32IN(I_L2WHH), 128);
  EW(add_bias_kernel, 256, bg0, F32IN(I_L0BIH), F32IN(I_L0BHH), 256);
  EW(add_bias_kernel, 256, bg1, F32IN(I_L1BIH), F32IN(I_L1BHH), 256);
  EW(add_bias_kernel, 256, bg2, F32IN(I_L2BIH), F32IN(I_L2BHH), 256);

  // ---- mlp1 (stats over B*T*N) ----
  EW(mlp1_fc1_kernel, (int)(NR1 * 64), F32IN(I_INPUTS), F32IN(I_M1W1), F32IN(I_M1B1), t1);
  GEMM(EPI_ELU_F32_STATS, t1, 64, Wm1b, 64, F32IN(I_M1B2), (int)NR1, 64, 64, 64, x1raw, (bf16*)nullptr, stats);
  finalize_stats_kernel<<<1, 64, 0, stream>>>(stats, mus, rss, 1.f / (float)NR1);
  EW(bn_write_kernel, (int)(NR1 * 64), x1raw, mus, rss, F32IN(I_M1G), F32IN(I_M1B),
     xn, 64, (bf16*)nullptr, 0, (bf16*)nullptr, 0, (float*)nullptr, (int)NR1);

  // ---- mlp2 global BN stats (over B*T*R, like the reference) ----
  for (int t = 0; t < T_; ++t) {
    EW(gather_x_kernel, MROWS * 128, xn, t, E);
    GEMM(EPI_ELU_BF16, E, 128, Wm2a, 128, F32IN(I_M2B1), MROWS, 64, 128, 64,
         (float*)nullptr, tmid, (float*)nullptr);
    GEMM(EPI_ELU_F32_STATS, tmid, 64, Wm2b, 64, F32IN(I_M2B2), MROWS, 64, 64, 64,
         raw, (bf16*)nullptr, stats2);
  }
  finalize_stats_kernel<<<1, 64, 0, stream>>>(stats2, mu2, rs2, 1.f / ((float)T_ * MROWS));

  bf16* Gs[3]  = {G0, G1, G2};
  int   ldG[3] = {128, 192, 192};
  bf16* Wgs[3] = {Wg0, Wg1, Wg2};
  int   Kg[3]  = {128, 192, 192};
  float* bgs[3] = {bg0, bg1, bg2};
  float* cs[3]  = {c0, c1, c2};

  // ---- recurrent scan ----
  for (int t = 0; t < T_; ++t) {
    // xt = BN(mlp2(node2edge(xn_t))) -> G0[:,0:64], G1[:,0:64]
    EW(gather_x_kernel, MROWS * 128, xn, t, E);
    GEMM(EPI_ELU_BF16, E, 128, Wm2a, 128, F32IN(I_M2B1), MROWS, 64, 128, 64,
         (float*)nullptr, tmid, (float*)nullptr);
    GEMM(EPI_ELU_F32, tmid, 64, Wm2b, 64, F32IN(I_M2B2), MROWS, 64, 64, 64,
         raw, (bf16*)nullptr, (float*)nullptr);
    EW(bn_write_kernel, MROWS * 64, raw, mu2, rs2, F32IN(I_M2G), F32IN(I_M2B),
       G0, 128, G1, 192, (bf16*)nullptr, 0, (float*)nullptr, MROWS);

    for (int L = 0; L < 3; ++L) {
      // fused LSTM gates: [x|h] @ [wih|whh]^T + (bih+bhh)
      GEMM(EPI_F32, Gs[L], ldG[L], Wgs[L], Kg[L], bgs[L], MROWS, 256, Kg[L], 256,
           gates, (bf16*)nullptr, (float*)nullptr);
      EW(lstm_cell_kernel, MROWS * 64, gates, cs[L], hl);
      // mlp3 on node aggregation (512 rows)
      EW(edge2node_kernel, B_ * N_ * 64, hl, node_in);
      GEMM(EPI_ELU_BF16, node_in, 64, Wm3a, 64, F32IN(I_M3B1), B_ * N_, 64, 64, 64,
           (float*)nullptr, tmid, (float*)nullptr);
      GEMM(EPI_ELU_F32_STATS, tmid, 64, Wm3b, 64, F32IN(I_M3B2), B_ * N_, 64, 64, 64,
           n3raw, (bf16*)nullptr, stats);
      finalize_stats_kernel<<<1, 64, 0, stream>>>(stats, mus, rss, 1.f / (float)(B_ * N_));
      // mlp4 on node2edge(BN(mlp3)) (32256 rows)
      EW(gather_bn_kernel, MROWS * 128, n3raw, mus, rss, F32IN(I_M3G), F32IN(I_M3B), E);
      GEMM(EPI_ELU_BF16, E, 128, Wm4a, 128, F32IN(I_M4B1), MROWS, 64, 128, 64,
           (float*)nullptr, tmid, (float*)nullptr);
      GEMM(EPI_ELU_F32_STATS, tmid, 64, Wm4b, 64, F32IN(I_M4B2), MROWS, 64, 64, 64,
           raw, (bf16*)nullptr, stats);
      finalize_stats_kernel<<<1, 64, 0, stream>>>(stats, mus, rss, 1.f / (float)MROWS);
      // h_new fan-out: next-step state slot + this-step concat slots
      bf16 *d0, *d1, *d2; int l0, l1, l2; float* df;
      if (L == 0) { d0 = G0 + 64; l0 = 128; d1 = G1 + 64;  l1 = 192; d2 = G2;      l2 = 192; df = nullptr; }
      else if (L == 1) { d0 = G1 + 128; l0 = 192; d1 = G2 + 64; l1 = 192; d2 = nullptr; l2 = 0; df = nullptr; }
      else { d0 = G2 + 128; l0 = 192; d1 = nullptr; l1 = 0; d2 = nullptr; l2 = 0; df = h2; }
      EW(bn_write_kernel, MROWS * 64, raw, mus, rss, F32IN(I_M4G), F32IN(I_M4B),
         d0, l0, d1, l1, d2, l2, df, MROWS);
    }
  }

  // ---- output projection: hs[2] @ out_w^T + out_b -> [B,R,E] ----
  EW(out_kernel, MROWS * 2, h2, F32IN(I_OUTW), F32IN(I_OUTB), (float*)d_out);
}
